// MRNN_83811991814685
// MI455X (gfx1250) — matrix operand order repaired and checked by reading, not compiled, not run on hardware
//
#include <hip/hip_runtime.h>
#include <hip/hip_bf16.h>
#include <stdint.h>

#define T_SEQ   40
#define HID     400
#define IN_DIM  814
#define FEAT    812
#define BATCH   2048
#define VAR_IDX 811
#define K_IN    832   // IN_DIM padded to 32
#define K_HH    416   // HID padded to 32
#define TILE_B  32
#define NHT     25    // 400/16 hid tiles
#define NWAVE   8
#define MAX_HT  4     // ceil(25/8)

typedef __attribute__((ext_vector_type(16))) __bf16 bf16x16;
typedef __attribute__((ext_vector_type(8)))  float  f32x8;

__device__ __forceinline__ unsigned short f2bf(float f) {
    union { float f; uint32_t u; } v; v.f = f;
    uint32_t r = v.u + 0x7FFF + ((v.u >> 16) & 1);   // round-to-nearest-even
    return (unsigned short)(r >> 16);
}

// A fragment: 16x32 bf16, M = lane&15, K chunks at 8*hi and 16+8*hi (ISA 7.12.2)
// Two 16-byte aligned chunks -> 2x ds_load_b128.
__device__ __forceinline__ bf16x16 load_a(const unsigned short* base, int stride,
                                          int k0, int lane) {
    const int m = lane & 15, hi = lane >> 4;
    const uint4* p0 = (const uint4*)(base + m * stride + k0 + 8 * hi);
    const uint4* p1 = (const uint4*)(base + m * stride + k0 + 16 + 8 * hi);
    union { bf16x16 v; uint4 q[2]; } u;
    u.q[0] = *p0;
    u.q[1] = *p1;
    return u.v;
}

// B fragment: 32x16 bf16 (B[k][n] = W[n0+n][k]); N = lane&15, K chunks at 8*hi and 16+8*hi.
// Two 16-byte aligned chunks per lane -> 2x global_load_b128 (clauses well).
__device__ __forceinline__ bf16x16 load_b(const unsigned short* wbase, int stride,
                                          int k0, int lane) {
    const int n = lane & 15, hi = lane >> 4;
    const uint4* p = (const uint4*)(wbase + n * stride + k0 + 8 * hi);
    union { bf16x16 v; uint4 q[2]; } u;
    u.q[0] = p[0];
    u.q[1] = p[2];
    return u.v;
}

__device__ __forceinline__ f32x8 wmma_bf16(bf16x16 a, bf16x16 b, f32x8 c) {
    return __builtin_amdgcn_wmma_f32_16x16x32_bf16(false, a, false, b,
                                                   (short)0, c, false, false);
}

__device__ __forceinline__ float sigf(float x) { return 1.0f / (1.0f + __expf(-x)); }

// ---------------------------------------------------------------------------
// Prep: convert weights to zero-padded bf16 (row-major [4H][K_pad]), zero xh,
// zero loss accumulator.
// ---------------------------------------------------------------------------
__global__ void mrnn_prep_kernel(const float* __restrict__ Wih_f,
                                 const float* __restrict__ Whh_f,
                                 unsigned short* __restrict__ Wih,
                                 unsigned short* __restrict__ Whh,
                                 float* __restrict__ xh,
                                 float* __restrict__ d_out) {
    const size_t tid    = (size_t)blockIdx.x * blockDim.x + threadIdx.x;
    const size_t stride = (size_t)gridDim.x * blockDim.x;
    for (size_t i = tid; i < (size_t)4 * HID * K_IN; i += stride) {
        const int row = (int)(i / K_IN), k = (int)(i % K_IN);
        Wih[i] = (k < IN_DIM) ? f2bf(Wih_f[(size_t)row * IN_DIM + k]) : (unsigned short)0;
    }
    for (size_t i = tid; i < (size_t)4 * HID * K_HH; i += stride) {
        const int row = (int)(i / K_HH), k = (int)(i % K_HH);
        Whh[i] = (k < HID) ? f2bf(Whh_f[(size_t)row * HID + k]) : (unsigned short)0;
    }
    for (size_t i = tid; i < (size_t)T_SEQ * BATCH; i += stride) xh[i] = 0.0f;
    if (tid == 0) d_out[0] = 0.0f;
}

// ---------------------------------------------------------------------------
// Main: one block per (batch tile, direction). 8 waves; each wave owns a
// stripe of the 25 hid-tiles and processes BOTH 16-row M-tiles of the batch
// tile with the same B (weight) fragments -> 8 WMMAs per 4 B-fragment loads.
// Cell state c stays in VGPR fragments across all 40 steps; h double-buffered
// bf16 in LDS; w_reg dot fused via shfl+atomics. Next-step x rows prefetched
// (global_prefetch) during compute to hide HBM latency of the staging phase.
// ---------------------------------------------------------------------------
__global__ __launch_bounds__(256) void mrnn_step_kernel(
    const float* __restrict__ values, const float* __restrict__ masks,
    const float* __restrict__ decay,  const float* __restrict__ rdecay,
    const float* __restrict__ b_ih,   const float* __restrict__ b_hh,
    const float* __restrict__ w_reg,
    const unsigned short* __restrict__ Wih,
    const unsigned short* __restrict__ Whh,
    float* __restrict__ xh) {
    extern __shared__ unsigned short smem[];
    unsigned short* lds_x = smem;                     // [TILE_B][K_IN]
    unsigned short* lds_h = smem + TILE_B * K_IN;     // [2][TILE_B][K_HH]

    const int dir  = blockIdx.y;
    const int b0   = blockIdx.x * TILE_B;
    const int tid  = threadIdx.x;
    const int lane = tid & 31;
    const int wid  = tid >> 5;      // hid-tile stripe 0..7
    const float* dec = dir ? rdecay : decay;

    // init: zero both h buffers (incl. K padding) and the x K-pad columns
    for (int i = tid; i < 2 * TILE_B * K_HH; i += 256) lds_h[i] = 0;
    for (int r = 0; r < TILE_B; ++r)
        for (int c = IN_DIM + tid; c < K_IN; c += 256) lds_x[r * K_IN + c] = 0;

    // persistent cell state: [ht][m-tile]
    f32x8 cst[MAX_HT][2];
#pragma unroll
    for (int i = 0; i < MAX_HT; ++i)
#pragma unroll
        for (int m = 0; m < 2; ++m)
#pragma unroll
            for (int e = 0; e < 8; ++e) cst[i][m][e] = 0.0f;

    const int hi = lane >> 4;
    const int nl = lane & 15;

    // hoist per-hid-tile biases and w_reg (constant across steps)
    float cbi[MAX_HT], cbf[MAX_HT], cbg[MAX_HT], cbo[MAX_HT], cwr[MAX_HT];
#pragma unroll
    for (int idx = 0; idx < MAX_HT; ++idx) {
        const int ht = wid + idx * NWAVE;
        const int n  = (ht < NHT ? ht : 0) * 16 + nl;
        cbi[idx] = b_ih[n]           + b_hh[n];
        cbf[idx] = b_ih[HID + n]     + b_hh[HID + n];
        cbg[idx] = b_ih[2 * HID + n] + b_hh[2 * HID + n];
        cbo[idx] = b_ih[3 * HID + n] + b_hh[3 * HID + n];
        cwr[idx] = w_reg[dir * HID + n];
    }

    for (int s = 0; s < T_SEQ; ++s) {
        const int t = dir ? (T_SEQ - 1 - s) : s;
        __syncthreads();   // prior step done reading lds_x / writing h
        // stage x_t tile (fp32 -> bf16)
        for (int r = 0; r < TILE_B; ++r) {
            const float* src = values + ((size_t)(b0 + r) * T_SEQ + t) * FEAT;
            for (int c = tid; c < FEAT; c += 256) lds_x[r * K_IN + c] = f2bf(src[c]);
        }
        if (tid < TILE_B) {
            const int b = b0 + tid;
            lds_x[tid * K_IN + FEAT]     = f2bf(masks[(size_t)b * T_SEQ + t]);
            lds_x[tid * K_IN + FEAT + 1] = f2bf(dec[(size_t)b * T_SEQ + t]);
        }
        __syncthreads();

        // prefetch next timestep's rows into cache during compute
        // (812 floats/row = 26 x 128B lines; ~3-4 global_prefetch per thread)
        if (s + 1 < T_SEQ) {
            const int tn = dir ? (t - 1) : (t + 1);
            for (int i = tid; i < TILE_B * 26; i += 256) {
                const int r = i / 26, line = i % 26;
                const char* p = (const char*)(values +
                    ((size_t)(b0 + r) * T_SEQ + tn) * FEAT) + (size_t)line * 128;
                __builtin_prefetch(p, 0, 3);
            }
        }

        const int cur = s & 1, nxt = cur ^ 1;
        const unsigned short* hbuf = lds_h + cur * TILE_B * K_HH;
        unsigned short*       hout = lds_h + nxt * TILE_B * K_HH;
        const int tout = dir ? (t - 1) : (t + 1);
        const bool emit = (tout >= 0) && (tout < T_SEQ);

        for (int idx = 0; idx < MAX_HT; ++idx) {
            const int ht = wid + idx * NWAVE;
            if (ht >= NHT) break;
            const int n0 = ht * 16;

            f32x8 acc[2][4];   // [m-tile][gate]
#pragma unroll
            for (int m = 0; m < 2; ++m)
#pragma unroll
                for (int g = 0; g < 4; ++g)
#pragma unroll
                    for (int e = 0; e < 8; ++e) acc[m][g][e] = 0.0f;

            // ---- input GEMM: x_t[32,832] x Wih^T slice ----
            const unsigned short* wbase = Wih + (size_t)n0 * K_IN;
            for (int k0 = 0; k0 < K_IN; k0 += 32) {
                bf16x16 a0 = load_a(lds_x, K_IN, k0, lane);
                bf16x16 a1 = load_a(lds_x + 16 * K_IN, K_IN, k0, lane);
#pragma unroll
                for (int g = 0; g < 4; ++g) {
                    bf16x16 b = load_b(wbase + (size_t)g * HID * K_IN, K_IN, k0, lane);
                    acc[0][g] = wmma_bf16(a0, b, acc[0][g]);
                    acc[1][g] = wmma_bf16(a1, b, acc[1][g]);
                }
            }
            // ---- hidden GEMM: h[32,416] x Whh^T slice ----
            const unsigned short* ubase = Whh + (size_t)n0 * K_HH;
            for (int k0 = 0; k0 < K_HH; k0 += 32) {
                bf16x16 a0 = load_a(hbuf, K_HH, k0, lane);
                bf16x16 a1 = load_a(hbuf + 16 * K_HH, K_HH, k0, lane);
#pragma unroll
                for (int g = 0; g < 4; ++g) {
                    bf16x16 b = load_b(ubase + (size_t)g * HID * K_HH, K_HH, k0, lane);
                    acc[0][g] = wmma_bf16(a0, b, acc[0][g]);
                    acc[1][g] = wmma_bf16(a1, b, acc[1][g]);
                }
            }

            // ---- LSTM elementwise (D layout: n=lane&15, m=e+8*hi) ----
            const int n = n0 + nl;
#pragma unroll
            for (int m = 0; m < 2; ++m) {
                f32x8 c = cst[idx][m];
                float hv[8];
#pragma unroll
                for (int e = 0; e < 8; ++e) {
                    const float ig = sigf(acc[m][0][e] + cbi[idx]);
                    const float fg = sigf(acc[m][1][e] + cbf[idx]);
                    const float gg = tanhf(acc[m][2][e] + cbg[idx]);
                    const float og = sigf(acc[m][3][e] + cbo[idx]);
                    const float cn = fg * c[e] + ig * gg;
                    c[e]  = cn;
                    hv[e] = og * tanhf(cn);
                }
                cst[idx][m] = c;

                // publish h_new into the other buffer
#pragma unroll
                for (int e = 0; e < 8; ++e) {
                    const int row = m * 16 + e + 8 * hi;
                    hout[row * K_HH + n] = f2bf(hv[e]);
                }

                // fused w_reg dot: hs[t+1] (fwd) / hs[t-1] (bwd) contribution
                if (emit) {
#pragma unroll
                    for (int e = 0; e < 8; ++e) {
                        float v = hv[e] * cwr[idx];
                        for (int o = 1; o < 16; o <<= 1) v += __shfl_xor(v, o, 16);
                        if (nl == 0)
                            atomicAdd(&xh[(size_t)tout * BATCH + b0 + m * 16 + e + 8 * hi], v);
                    }
                }
            }
        }
    }
}

// ---------------------------------------------------------------------------
// Loss + imputations: one block per timestep.
// ---------------------------------------------------------------------------
__global__ __launch_bounds__(256) void mrnn_loss_kernel(
    const float* __restrict__ values, const float* __restrict__ masks,
    const float* __restrict__ xh, const float* __restrict__ b_reg,
    float* __restrict__ d_out) {
    const int t = blockIdx.x;
    const float br = b_reg[0];
    float num = 0.0f, den = 0.0f;
    for (int b = threadIdx.x; b < BATCH; b += 256) {
        const float xv  = values[((size_t)b * T_SEQ + t) * FEAT + VAR_IDX];
        const float m   = masks[(size_t)b * T_SEQ + t];
        const float xhv = xh[(size_t)t * BATCH + b] + br;
        d_out[1 + (size_t)b * T_SEQ + t] = xhv;   // imputations [B][T]
        num += fabsf(xv - xhv) * m;
        den += m;
    }
    __shared__ float sn[256], sd[256];
    sn[threadIdx.x] = num; sd[threadIdx.x] = den;
    __syncthreads();
    for (int o = 128; o > 0; o >>= 1) {
        if (threadIdx.x < o) {
            sn[threadIdx.x] += sn[threadIdx.x + o];
            sd[threadIdx.x] += sd[threadIdx.x + o];
        }
        __syncthreads();
    }
    if (threadIdx.x == 0)
        atomicAdd(d_out, (sn[0] / (sd[0] + 1e-5f)) / (float)T_SEQ);
}

extern "C" void kernel_launch(void* const* d_in, const int* in_sizes, int n_in,
                              void* d_out, int out_size, void* d_ws, size_t ws_size,
                              hipStream_t stream) {
    const float* values = (const float*)d_in[0];
    const float* masks  = (const float*)d_in[1];
    const float* decay  = (const float*)d_in[2];
    const float* rdecay = (const float*)d_in[3];
    const float* W_ih   = (const float*)d_in[4];
    const float* W_hh   = (const float*)d_in[5];
    const float* b_ih   = (const float*)d_in[6];
    const float* b_hh   = (const float*)d_in[7];
    const float* w_reg  = (const float*)d_in[8];
    const float* b_reg  = (const float*)d_in[9];
    float* out = (float*)d_out;

    char* ws = (char*)d_ws;
    const size_t wih_bytes = (size_t)4 * HID * K_IN * sizeof(unsigned short); // 2.66 MB
    const size_t whh_bytes = (size_t)4 * HID * K_HH * sizeof(unsigned short); // 1.33 MB
    unsigned short* Wih_bf = (unsigned short*)ws;
    unsigned short* Whh_bf = (unsigned short*)(ws + wih_bytes);
    float*          xh     = (float*)(ws + wih_bytes + whh_bytes);            // T*B f32

    mrnn_prep_kernel<<<512, 256, 0, stream>>>(W_ih, W_hh, Wih_bf, Whh_bf, xh, out);

    const size_t lds_bytes = (size_t)(TILE_B * K_IN + 2 * TILE_B * K_HH) * sizeof(unsigned short);
    dim3 grid(BATCH / TILE_B, 2);
    mrnn_step_kernel<<<grid, 256, lds_bytes, stream>>>(
        values, masks, decay, rdecay, b_ih, b_hh, w_reg, Wih_bf, Whh_bf, xh);

    mrnn_loss_kernel<<<T_SEQ, 256, 0, stream>>>(values, masks, xh, b_reg, out);
}
